// GCNEmbedding_20684562498293
// MI455X (gfx1250) — compile-verified, matrix operand order checked by reading, not run on hardware
//
#include <hip/hip_runtime.h>
#include <hip/hip_bf16.h>

#define N_CH 128
#define TILES_PER_BLOCK 10   // 160 rows per block; 100000 / 160 = 625 exact

typedef __attribute__((ext_vector_type(16))) __bf16 bf16x16;
typedef __attribute__((ext_vector_type(8)))  __bf16 bf16x8;
typedef __attribute__((ext_vector_type(4)))  __bf16 bf16x4;
typedef __attribute__((ext_vector_type(8)))  float  f32x8;

__device__ __forceinline__ void gaddf(float* p, float v) {
    unsafeAtomicAdd(p, v);   // lowers to global_atomic_add_f32
}

// ---------------- degree / normalization ----------------
__global__ __launch_bounds__(256) void gcn_deg_init(float* deg, int n) {
    int i = blockIdx.x * 256 + threadIdx.x;
    if (i < n) deg[i] = 1.0f;                 // self loop
}

__global__ __launch_bounds__(256) void gcn_deg_edges(const int* __restrict__ dst, float* deg, int e) {
    int i = blockIdx.x * 256 + threadIdx.x;
    if (i < e) gaddf(&deg[dst[i]], 1.0f);
}

__global__ __launch_bounds__(256) void gcn_deg_rsqrt(float* deg, int n) {
    int i = blockIdx.x * 256 + threadIdx.x;
    if (i < n) deg[i] = rsqrtf(deg[i]);       // in-place: deg -> dis
}

// ---------------- fused GEMM (WMMA bf16) ----------------
// Block computes TILES_PER_BLOCK row-tiles of 16 rows x 128 cols, staging W
// into LDS once. A tiles are ping-pong buffered: one barrier per tile.
// FUSED=0: A' = A (raw features).
// FUSED=1: A' = relu(dis[r]*A[r][c] + bias[c])  (layer-1 epilogue fused in).
// Writes hs = dis[r] * (A' @ W) and acc = hs (self-loop init for scatter).
template <int FUSED>
__global__ __launch_bounds__(256) void gcn_gemm(const float* __restrict__ A,
                                                const float* __restrict__ W,    // 128x128 row-major (K x N)
                                                const float* __restrict__ bias,
                                                const float* __restrict__ dis,
                                                float* __restrict__ hs,
                                                float* __restrict__ acc) {
    __shared__ __align__(16) __bf16 Bs[128 * 136];
    __shared__ __align__(16) __bf16 As[2][16 * 136];
    __shared__ float sdis[2][16];

    const int t = threadIdx.x;

    // Stage W transposed once per block: Bs[n*136 + k] = bf16(W[k*128 + n])
    #pragma unroll
    for (int i = 0; i < 16; ++i) {
        int idx4 = t + i * 256;               // 0..4095 float4s
        int k  = idx4 >> 5;                   // 0..127
        int n4 = (idx4 & 31) * 4;
        float4 v = *(const float4*)&W[k * 128 + n4];
        Bs[(n4 + 0) * 136 + k] = (__bf16)v.x;
        Bs[(n4 + 1) * 136 + k] = (__bf16)v.y;
        Bs[(n4 + 2) * 136 + k] = (__bf16)v.z;
        Bs[(n4 + 3) * 136 + k] = (__bf16)v.w;
    }

    const int w    = t >> 5;                  // wave 0..7 -> output cols [w*16, w*16+16)
    const int lane = t & 31;
    const int hi   = lane >> 4;               // half-wave select (K lo/hi 8)
    const int lr   = lane & 15;               // A row / B col within tile
    const int col  = w * 16 + lr;
    const int tile0 = blockIdx.x * TILES_PER_BLOCK;

    for (int it = 0; it < TILES_PER_BLOCK; ++it) {
        const int rowbase = (tile0 + it) * 16;
        const int buf = it & 1;
        __bf16* Ab = &As[buf][0];

        // Stage A tile (16 x 128) as bf16, optionally fused bias+relu+dis
        #pragma unroll
        for (int i = 0; i < 2; ++i) {
            int idx4 = t + i * 256;           // 0..511 float4s
            int r  = idx4 >> 5;
            int c4 = (idx4 & 31) * 4;
            float4 v = *(const float4*)&A[(size_t)(rowbase + r) * 128 + c4];
            if (FUSED) {
                float ds = dis[rowbase + r];
                v.x = fmaxf(v.x * ds + bias[c4 + 0], 0.0f);
                v.y = fmaxf(v.y * ds + bias[c4 + 1], 0.0f);
                v.z = fmaxf(v.z * ds + bias[c4 + 2], 0.0f);
                v.w = fmaxf(v.w * ds + bias[c4 + 3], 0.0f);
            }
            bf16x4 pk = { (__bf16)v.x, (__bf16)v.y, (__bf16)v.z, (__bf16)v.w };
            *(bf16x4*)&Ab[r * 136 + c4] = pk;
        }
        if (t < 16) sdis[buf][t] = dis[rowbase + t];
        __syncthreads();   // A[buf] (and W on first iter) visible; also orders
                           // WAR reuse of A[buf] from two iterations ago

        f32x8 c = {};
        #pragma unroll
        for (int kb = 0; kb < 128; kb += 32) {
            bf16x16 a, b;
            const __bf16* ap = &Ab[lr * 136 + kb + hi * 8];
            const __bf16* bp = &Bs[col * 136 + kb + hi * 8];
            ((bf16x8*)&a)[0] = *(const bf16x8*)(ap);       // K = kb + hi*8 .. +7
            ((bf16x8*)&a)[1] = *(const bf16x8*)(ap + 16);  // K = kb+16 + hi*8 .. +7
            ((bf16x8*)&b)[0] = *(const bf16x8*)(bp);
            ((bf16x8*)&b)[1] = *(const bf16x8*)(bp + 16);
            c = __builtin_amdgcn_wmma_f32_16x16x32_bf16(false, a, false, b, (short)0, c, false, false);
        }

        // C layout: VGPR j, lanes 0-15 -> row j, lanes 16-31 -> row j+8
        #pragma unroll
        for (int j = 0; j < 8; ++j) {
            int r = j + hi * 8;
            float v = c[j] * sdis[buf][r];
            size_t off = (size_t)(rowbase + r) * 128 + col;
            hs[off]  = v;
            acc[off] = v;                      // self-loop contribution
        }
    }
}

// ---------------- edge scatter-add ----------------
// One wave per edge: 32 lanes x float4 = 128 channels.
__global__ __launch_bounds__(256) void gcn_scatter(const float* __restrict__ hs,
                                                   const int* __restrict__ src,
                                                   const int* __restrict__ dst,
                                                   float* __restrict__ acc, int e) {
    int wave = threadIdx.x >> 5;
    int lane = threadIdx.x & 31;
    int ei = blockIdx.x * 8 + wave;
    if (ei >= e) return;
    int s = src[ei];
    int d = dst[ei];
    const float4 v = *(const float4*)&hs[(size_t)s * 128 + lane * 4];
    float* base = &acc[(size_t)d * 128 + lane * 4];
    gaddf(base + 0, v.x);
    gaddf(base + 1, v.y);
    gaddf(base + 2, v.z);
    gaddf(base + 3, v.w);
}

// ---------------- final epilogue: out = dis[row]*acc + b2 ----------------
__global__ __launch_bounds__(256) void gcn_finalize(float* __restrict__ out,
                                                    const float* __restrict__ dis,
                                                    const float* __restrict__ bias, int total) {
    int i = blockIdx.x * 256 + threadIdx.x;
    if (i < total) {
        int row = i >> 7;
        int ch  = i & 127;
        out[i] = dis[row] * out[i] + bias[ch];
    }
}

extern "C" void kernel_launch(void* const* d_in, const int* in_sizes, int n_in,
                              void* d_out, int out_size, void* d_ws, size_t ws_size,
                              hipStream_t stream) {
    const float* x  = (const float*)d_in[0];
    const int*   ei = (const int*)d_in[2];      // (2, E) int32
    const float* W1 = (const float*)d_in[5];
    const float* b1 = (const float*)d_in[6];
    const float* W2 = (const float*)d_in[7];
    const float* b2 = (const float*)d_in[8];

    const int N = in_sizes[0] / N_CH;           // 100000
    const int E = in_sizes[2] / 2;              // 1600000
    const int* src = ei;
    const int* dst = ei + E;

    float* dis  = (float*)d_ws;                 // N
    float* hs   = dis + N;                      // N*128 (reused for both layers)
    float* acc1 = hs + (size_t)N * N_CH;        // N*128
    float* out  = (float*)d_out;                // N*128, doubles as layer-2 accumulator

    const int nb_n   = (N + 255) / 256;
    const int nb_e   = (E + 255) / 256;
    const int nb_gm  = N / (16 * TILES_PER_BLOCK);   // 625 exact
    const int nb_sc  = (E + 7) / 8;
    const int nb_fin = (N * N_CH + 255) / 256;

    // normalization: deg -> rsqrt (in place, buffer 'dis')
    gcn_deg_init <<<nb_n, 256, 0, stream>>>(dis, N);
    gcn_deg_edges<<<nb_e, 256, 0, stream>>>(dst, dis, E);
    gcn_deg_rsqrt<<<nb_n, 256, 0, stream>>>(dis, N);

    // layer 1: hs1 = dis .* (x @ W1), acc1 initialized with self-loop term
    gcn_gemm<0><<<nb_gm, 256, 0, stream>>>(x, W1, b1, dis, hs, acc1);
    gcn_scatter<<<nb_sc, 256, 0, stream>>>(hs, src, dst, acc1, E);

    // layer 2: A = relu(dis .* acc1 + b1) fused; hs2 = dis .* (A @ W2); out init = hs2
    gcn_gemm<1><<<nb_gm, 256, 0, stream>>>(acc1, W2, b1, dis, hs, out);
    gcn_scatter<<<nb_sc, 256, 0, stream>>>(hs, src, dst, out, E);

    // out = dis .* out + b2 (in place)
    gcn_finalize<<<nb_fin, 256, 0, stream>>>(out, dis, b2, N * N_CH);
}
